// MOELayer_81990925680835
// MI455X (gfx1250) — matrix operand order repaired and checked by reading, not compiled, not run on hardware
//
#include <hip/hip_runtime.h>

#define N_TOKENS 65536
#define D_MODEL  256
#define D_HID    256
#define N_EXP    8

typedef __attribute__((ext_vector_type(16))) _Float16 v16h;
typedef __attribute__((ext_vector_type(8)))  _Float16 v8h;
typedef __attribute__((ext_vector_type(4)))  _Float16 v4h;
typedef __attribute__((ext_vector_type(8)))  float    v8f;

// ---------------- kernel 0: zero counters ----------------
__global__ void init_counts_kernel(int* counts) {
    if (threadIdx.x < N_EXP) counts[threadIdx.x] = 0;
}

// ---------------- kernel 1: convert + swizzle weights into WMMA B-fragment order ----
// B-fragment (32x16 KxN, f16): element(lane l, elem i) = B[K=(i>>3)*16+(l>>4)*8+(i&7)][N=l&15]
// fragment id within expert = nt*8 + kt (so the K-loop streams 8KB contiguously)
__global__ __launch_bounds__(256) void wswizzle_kernel(
    const float* __restrict__ W1, const float* __restrict__ W2,
    _Float16* __restrict__ W1h, _Float16* __restrict__ W2h)
{
    int idx = blockIdx.x * 256 + threadIdx.x;           // 0 .. 2*8*256*256-1
    const int EW = N_EXP * D_MODEL * D_HID;             // 524288
    const float* src = W1; _Float16* dst = W1h; int s = idx;
    if (idx >= EW) { src = W2; dst = W2h; s = idx - EW; }
    int e = s >> 16;
    int k = (s >> 8) & 255;
    int n = s & 255;
    float v = src[s];
    int kk = k & 31, kt = k >> 5, nt = n >> 4;
    int l = (n & 15) | (((kk >> 3) & 1) << 4);
    int i = (kk & 7) | ((kk >> 4) << 3);
    dst[e * 65536 + ((nt << 3) + kt) * 512 + l * 16 + i] = (_Float16)v;
}

// ---------------- kernel 2: gating (softmax top-1) ----------------
__global__ __launch_bounds__(256) void gate_kernel(
    const float* __restrict__ x, const float* __restrict__ wg,
    float* __restrict__ gate_val, int* __restrict__ assign_, int* __restrict__ counts)
{
    __shared__ float wgs[N_EXP * 257];   // padded stride -> conflict-free
    __shared__ float logits[256];
    __shared__ int   lcnt[N_EXP];

    int t = threadIdx.x;
    if (t < N_EXP) lcnt[t] = 0;
    for (int i = t; i < N_EXP * D_MODEL; i += 256)
        wgs[(i >> 8) * 257 + (i & 255)] = wg[i];
    __syncthreads();

    int tokLocal = t >> 3, e = t & 7;
    int token = blockIdx.x * 32 + tokLocal;
    const float4* xr = (const float4*)(x + (long)token * D_MODEL);
    const float*  wr = wgs + e * 257;
    float acc = 0.f;
#pragma unroll 8
    for (int k4 = 0; k4 < 64; ++k4) {
        float4 v = xr[k4];
        acc += v.x * wr[k4*4+0] + v.y * wr[k4*4+1] + v.z * wr[k4*4+2] + v.w * wr[k4*4+3];
    }
    logits[t] = acc;
    __syncthreads();

    if (t < 32) {
        int token2 = blockIdx.x * 32 + t;
        const float* lg = logits + t * 8;
        float lmax = lg[0]; int amax = 0;
#pragma unroll
        for (int q = 1; q < N_EXP; ++q) { float v = lg[q]; if (v > lmax) { lmax = v; amax = q; } }
        float s = 0.f;
#pragma unroll
        for (int q = 0; q < N_EXP; ++q) s += __expf(lg[q] - lmax);
        gate_val[token2] = 1.0f / s;         // == softmax max value
        assign_[token2]  = amax;
        atomicAdd(&lcnt[amax], 1);
    }
    __syncthreads();
    if (t < N_EXP) atomicAdd(&counts[t], lcnt[t]);
}

// ---------------- kernel 3: scan counts -> offsets / tile offsets / cursors ----
__global__ void scan_kernel(const int* __restrict__ counts,
                            int* offsets, int* tileOff, int* cursors)
{
    if (threadIdx.x == 0) {
        int off = 0, to = 0;
        for (int e = 0; e < N_EXP; ++e) {
            offsets[e] = off; tileOff[e] = to; cursors[e] = off;
            off += counts[e];
            to  += (counts[e] + 31) >> 5;    // M-tile = 32 tokens
        }
        offsets[N_EXP] = off; tileOff[N_EXP] = to;
    }
}

// ---------------- kernel 4: wave-aggregated scatter -> perm ----------------
__global__ __launch_bounds__(256) void scatter_kernel(
    const int* __restrict__ assign_, int* __restrict__ cursors, int* __restrict__ perm)
{
    int tid  = blockIdx.x * 256 + threadIdx.x;
    int lane = threadIdx.x & 31;
    int e = assign_[tid];
#pragma unroll
    for (int q = 0; q < N_EXP; ++q) {
        unsigned mask = (unsigned)__ballot(e == q);
        if (mask) {
            int leader = __ffs(mask) - 1;
            int base = 0;
            if (lane == leader) base = atomicAdd(&cursors[q], __popc(mask));
            base = __shfl(base, leader, 32);
            if (e == q) {
                int rank = __popc(mask & ((1u << lane) - 1u));
                perm[base + rank] = tid;
            }
        }
    }
}

// ---------------- kernel 5: grouped-FFN via WMMA f16 ----------------
__device__ inline void cvt4(v16h& av, int base, float4 c) {
    av[base+0] = (_Float16)c.x; av[base+1] = (_Float16)c.y;
    av[base+2] = (_Float16)c.z; av[base+3] = (_Float16)c.w;
}
// A-fragment (16x32 MxK f16): lane l, elem i -> M=l&15, K=(i/8)*16 + (l>>4)*8 + (i&7)
__device__ inline v16h load_a_f16(const _Float16* ap) {
    v8h lo = *(const v8h*)ap;
    v8h hi = *(const v8h*)(ap + 16);
    return __builtin_shufflevector(lo, hi, 0,1,2,3,4,5,6,7,8,9,10,11,12,13,14,15);
}
__device__ inline v16h load_a_f32(const float* ap) {
    float4 c0 = *(const float4*)(ap);
    float4 c1 = *(const float4*)(ap + 4);
    float4 c2 = *(const float4*)(ap + 16);
    float4 c3 = *(const float4*)(ap + 20);
    v16h av;
    cvt4(av, 0, c0); cvt4(av, 4, c1); cvt4(av, 8, c2); cvt4(av, 12, c3);
    return av;
}

#define XS_STRIDE 264   // f16 elems/row: capacity-optimal LDS banking for A reads
#define HS_STRIDE 268   // f32 elems/row

__global__ __launch_bounds__(256) void ffn_kernel(
    const float* __restrict__ x, const float* __restrict__ b1v, const float* __restrict__ b2v,
    const _Float16* __restrict__ W1h, const _Float16* __restrict__ W2h,
    const int* __restrict__ perm, const float* __restrict__ gate_val,
    const int* __restrict__ offsets, const int* __restrict__ tileOff,
    float* __restrict__ out)
{
    __shared__ _Float16 xs[32 * XS_STRIDE];
    __shared__ float    hs[32 * HS_STRIDE];
    __shared__ int      toks[32];
    __shared__ float    gvs[32];

    int b = blockIdx.x;
    if (b >= tileOff[N_EXP]) return;
    int e = 0;
    while (b >= tileOff[e + 1]) ++e;                    // uniform per block
    int rowBase = offsets[e] + (b - tileOff[e]) * 32;
    int valid = offsets[e + 1] - rowBase; if (valid > 32) valid = 32;

    int t = threadIdx.x;
    if (t < 32) {
        int tk = perm[rowBase + (t < valid ? t : valid - 1)];
        toks[t] = tk;
        gvs[t]  = gate_val[tk];
    }
    // stage gathered x rows -> LDS as f16 (coalesced 128B chunks per 8 threads)
    {
        int r = t >> 3, part = t & 7;
        int tk = perm[rowBase + (r < valid ? r : valid - 1)];
        const float4* src = (const float4*)(x + (long)tk * D_MODEL);
        _Float16* dst = xs + r * XS_STRIDE;
#pragma unroll
        for (int it = 0; it < 8; ++it) {
            float4 v = src[part + it * 8];
            v4h p; p[0]=(_Float16)v.x; p[1]=(_Float16)v.y; p[2]=(_Float16)v.z; p[3]=(_Float16)v.w;
            *(v4h*)(dst + (part + it * 8) * 4) = p;
        }
    }
    __syncthreads();

    int lane = t & 31, wave = t >> 5;
    int lrow = lane & 15, lhi = lane >> 4;

    // ---- GEMM1: h[32x256] = x[32x256] @ W1[e]  (f16 in, f32 acc) ----
    v8f acc[2][2] = {};
    const _Float16* wb1 = W1h + e * 65536;
#pragma unroll
    for (int kt = 0; kt < 8; ++kt) {
        v16h a0 = load_a_f16(xs + lrow * XS_STRIDE + kt * 32 + lhi * 8);
        v16h a1 = load_a_f16(xs + (16 + lrow) * XS_STRIDE + kt * 32 + lhi * 8);
#pragma unroll
        for (int ntl = 0; ntl < 2; ++ntl) {
            int nt = wave * 2 + ntl;
            v16h bf = *(const v16h*)(wb1 + ((nt << 3) + kt) * 512 + lane * 16);
            acc[0][ntl] = __builtin_amdgcn_wmma_f32_16x16x32_f16(false, a0, false, bf, (short)0, acc[0][ntl], false, false);
            acc[1][ntl] = __builtin_amdgcn_wmma_f32_16x16x32_f16(false, a1, false, bf, (short)0, acc[1][ntl], false, false);
        }
    }
    // epilogue 1: +b1, ReLU, store fp32 to LDS (C layout: m = mt*16 + lhi*8 + r, n = nt*16 + lrow)
#pragma unroll
    for (int ntl = 0; ntl < 2; ++ntl) {
        int n = (wave * 2 + ntl) * 16 + lrow;
        float bias = b1v[e * D_HID + n];
#pragma unroll
        for (int mt = 0; mt < 2; ++mt)
#pragma unroll
            for (int r = 0; r < 8; ++r) {
                int m = mt * 16 + lhi * 8 + r;
                float hv = acc[mt][ntl][r] + bias;
                hs[m * HS_STRIDE + n] = hv > 0.f ? hv : 0.f;
            }
    }
    __syncthreads();

    // ---- GEMM2: y[32x256] = h @ W2[e] ----
    v8f acc2[2][2] = {};
    const _Float16* wb2 = W2h + e * 65536;
#pragma unroll
    for (int kt = 0; kt < 8; ++kt) {
        v16h a0 = load_a_f32(hs + lrow * HS_STRIDE + kt * 32 + lhi * 8);
        v16h a1 = load_a_f32(hs + (16 + lrow) * HS_STRIDE + kt * 32 + lhi * 8);
#pragma unroll
        for (int ntl = 0; ntl < 2; ++ntl) {
            int nt = wave * 2 + ntl;
            v16h bf = *(const v16h*)(wb2 + ((nt << 3) + kt) * 512 + lane * 16);
            acc2[0][ntl] = __builtin_amdgcn_wmma_f32_16x16x32_f16(false, a0, false, bf, (short)0, acc2[0][ntl], false, false);
            acc2[1][ntl] = __builtin_amdgcn_wmma_f32_16x16x32_f16(false, a1, false, bf, (short)0, acc2[1][ntl], false, false);
        }
    }
    // epilogue 2: +b2, scale by gate, scatter rows (mask padded rows)
#pragma unroll
    for (int mt = 0; mt < 2; ++mt) {
        int mb = mt * 16 + lhi * 8;
        int tkr[8]; float gvr[8];
#pragma unroll
        for (int r = 0; r < 8; ++r) { tkr[r] = toks[mb + r]; gvr[r] = gvs[mb + r]; }
#pragma unroll
        for (int ntl = 0; ntl < 2; ++ntl) {
            int n = (wave * 2 + ntl) * 16 + lrow;
            float bias = b2v[e * D_MODEL + n];
#pragma unroll
            for (int r = 0; r < 8; ++r) {
                if (mb + r < valid)
                    out[(long)tkr[r] * D_MODEL + n] = (acc2[mt][ntl][r] + bias) * gvr[r];
            }
        }
    }
}

// ---------------- host launcher ----------------
extern "C" void kernel_launch(void* const* d_in, const int* in_sizes, int n_in,
                              void* d_out, int out_size, void* d_ws, size_t ws_size,
                              hipStream_t stream) {
    const float* x  = (const float*)d_in[0];
    const float* wg = (const float*)d_in[1];
    const float* W1 = (const float*)d_in[2];
    const float* b1 = (const float*)d_in[3];
    const float* W2 = (const float*)d_in[4];
    const float* b2 = (const float*)d_in[5];
    float* out = (float*)d_out;

    char* ws = (char*)d_ws;
    const size_t N4 = (size_t)N_TOKENS * 4;
    int*   counts   = (int*)(ws + 0);
    int*   cursors  = (int*)(ws + 64);
    int*   offsets  = (int*)(ws + 128);
    int*   tileOff  = (int*)(ws + 192);
    float* gate_val = (float*)(ws + 256);
    int*   assign_  = (int*)(ws + 256 + N4);
    int*   perm     = (int*)(ws + 256 + 2 * N4);
    _Float16* W1h   = (_Float16*)(ws + 256 + 3 * N4);          // 786688, 32B aligned
    _Float16* W2h   = W1h + (size_t)N_EXP * D_MODEL * D_HID;

    init_counts_kernel<<<1, 32, 0, stream>>>(counts);
    wswizzle_kernel<<<4096, 256, 0, stream>>>(W1, W2, W1h, W2h);
    gate_kernel<<<N_TOKENS / 32, 256, 0, stream>>>(x, wg, gate_val, assign_, counts);
    scan_kernel<<<1, 32, 0, stream>>>(counts, offsets, tileOff, cursors);
    scatter_kernel<<<N_TOKENS / 256, 256, 0, stream>>>(assign_, cursors, perm);
    ffn_kernel<<<N_TOKENS / 32 + N_EXP, 256, 0, stream>>>(
        x, b1, b2, W1h, W2h, perm, gate_val, offsets, tileOff, out);
}
